// Head_20684562497854
// MI455X (gfx1250) — matrix operand order repaired and checked by reading, not compiled, not run on hardware
//
#include <hip/hip_runtime.h>
#include <hip/hip_bf16.h>
#include <stdint.h>

#define B_ 8
#define T_ 2048
#define C_ 1024
#define H_ 64

typedef __attribute__((ext_vector_type(16))) __bf16 v16bf;
typedef __attribute__((ext_vector_type(8)))  float  v8f;

union FragBF {
    v16bf v;
    __bf16 h[16];
    uint4  q[2];
    unsigned short us[16];
};

static __device__ __forceinline__ uint4 ld16g(const unsigned short* p) {
    return *(const uint4*)p;
}

static __device__ __forceinline__ unsigned short f2bf(float f) {
    __bf16 b = (__bf16)f;
    return __builtin_bit_cast(unsigned short, b);
}

// ---------------------------------------------------------------------------
// Kernel 0: one-time W transpose + f32->bf16 convert.
// WT_m (H, C) bf16 so projection B-fragments are K-contiguous (2x b128 loads).
// ---------------------------------------------------------------------------
__global__ void __launch_bounds__(256) wprep_kernel(
    const float* __restrict__ Wq, const float* __restrict__ Wk,
    const float* __restrict__ Wv, unsigned short* __restrict__ wtq,
    unsigned short* __restrict__ wtk, unsigned short* __restrict__ wtv)
{
    const int idx = blockIdx.x * 256 + threadIdx.x;   // over H_*C_ elements
    const int h = idx >> 10;                          // C_ = 1024
    const int c = idx & 1023;
    const size_t src = (size_t)c * H_ + h;            // W is (C,H)
    const size_t dst = (size_t)h * C_ + c;            // WT is (H,C)
    wtq[dst] = f2bf(Wq[src]);
    wtk[dst] = f2bf(Wk[src]);
    wtv[dst] = f2bf(Wv[src]);
}

// ---------------------------------------------------------------------------
// Kernel 1: q/k/v projections.  x(B,T,C) f32 @ WT(H,C) bf16 -> bf16 q,k (T,H)
// and vT (H,T).  One 16-row tile per block; 4 waves = 4 N-tiles of H=64.
// ---------------------------------------------------------------------------
__global__ void __launch_bounds__(128) qkv_proj_kernel(
    const float* __restrict__ x,
    const unsigned short* __restrict__ wtq,
    const unsigned short* __restrict__ wtk,
    const unsigned short* __restrict__ wtv,
    unsigned short* __restrict__ qb, unsigned short* __restrict__ kb,
    unsigned short* __restrict__ vtb)
{
    const int lane = threadIdx.x & 31;
    const int wid  = threadIdx.x >> 5;        // N-tile 0..3
    const int lm   = lane & 15;
    const int hg   = lane >> 4;               // half-group of wave32
    const int b    = blockIdx.x >> 7;         // 128 row-tiles per batch
    const int t0   = (blockIdx.x & 127) << 4;

    const float* X   = x + ((size_t)b * T_ + t0) * C_;
    const int    col = wid * 16 + lm;

    // B-fragment row base: lane = col, elements are K-contiguous in WT
    const unsigned short* wq = wtq + (size_t)col * C_;
    const unsigned short* wk = wtk + (size_t)col * C_;
    const unsigned short* wv = wtv + (size_t)col * C_;

    v8f aq = {}; v8f ak = {}; v8f av = {};

    for (int kbase = 0; kbase < C_; kbase += 32) {
        // --- A fragment: 16x32 tile of x, f32 -> bf16 ---
        // lane = row lm; elements 0..15 -> c = kbase+16*hg+e
        FragBF xa;
        const float* xr = X + (size_t)lm * C_ + kbase;
        float4 f0 = *(const float4*)(xr + hg * 16);
        float4 f1 = *(const float4*)(xr + hg * 16 + 4);
        float4 f2 = *(const float4*)(xr + hg * 16 + 8);
        float4 f3 = *(const float4*)(xr + hg * 16 + 12);
        xa.h[0]=(__bf16)f0.x;  xa.h[1]=(__bf16)f0.y;  xa.h[2]=(__bf16)f0.z;  xa.h[3]=(__bf16)f0.w;
        xa.h[4]=(__bf16)f1.x;  xa.h[5]=(__bf16)f1.y;  xa.h[6]=(__bf16)f1.z;  xa.h[7]=(__bf16)f1.w;
        xa.h[8]=(__bf16)f2.x;  xa.h[9]=(__bf16)f2.y;  xa.h[10]=(__bf16)f2.z; xa.h[11]=(__bf16)f2.w;
        xa.h[12]=(__bf16)f3.x; xa.h[13]=(__bf16)f3.y; xa.h[14]=(__bf16)f3.z; xa.h[15]=(__bf16)f3.w;

        // --- B fragments: 32x16 of W via WT, K = kbase + 16*hg + e (contiguous) ---
        const int ko = kbase + hg * 16;
        FragBF wqf, wkf, wvf;
        wqf.q[0] = ld16g(wq + ko);  wqf.q[1] = ld16g(wq + ko + 8);
        wkf.q[0] = ld16g(wk + ko);  wkf.q[1] = ld16g(wk + ko + 8);
        wvf.q[0] = ld16g(wv + ko);  wvf.q[1] = ld16g(wv + ko + 8);

        aq = __builtin_amdgcn_wmma_f32_16x16x32_bf16(false, xa.v, false, wqf.v, (short)0, aq, false, false);
        ak = __builtin_amdgcn_wmma_f32_16x16x32_bf16(false, xa.v, false, wkf.v, (short)0, ak, false, false);
        av = __builtin_amdgcn_wmma_f32_16x16x32_bf16(false, xa.v, false, wvf.v, (short)0, av, false, false);
    }

    // C/D layout: VGPR r -> row r + 8*hg, lane%16 -> col
#pragma unroll
    for (int r = 0; r < 8; ++r) {
        const int trow = t0 + r + hg * 8;
        qb [((size_t)b * T_ + trow) * H_ + col] = f2bf(aq[r]);
        kb [((size_t)b * T_ + trow) * H_ + col] = f2bf(ak[r]);
        vtb[((size_t)b * H_ + col) * T_ + trow] = f2bf(av[r]);   // V transposed
    }
}

// ---------------------------------------------------------------------------
// Kernel 2: flash attention.  One wave per 16-query tile, key blocks of 32,
// online softmax in exp2 domain, P staged through per-wave LDS for D->A
// re-layout.  No block-wide barriers (waves have divergent trip counts).
// ---------------------------------------------------------------------------
__global__ void __launch_bounds__(128) attn_kernel(
    const unsigned short* __restrict__ qbuf,
    const unsigned short* __restrict__ kbuf,
    const unsigned short* __restrict__ vtbuf,
    float* __restrict__ out)
{
    __shared__ __align__(16) unsigned short plds[4][16 * 32];

    const int lane = threadIdx.x & 31;
    const int wid  = threadIdx.x >> 5;
    const int lm   = lane & 15;
    const int hg   = lane >> 4;
    const int tile = blockIdx.x * 4 + wid;
    const int b    = tile >> 7;
    const int t0   = (tile & 127) << 4;

    // --- Q A-fragments, held in registers for the whole tile ---
    FragBF qa0, qa1;
    {
        const unsigned short* qr = qbuf + ((size_t)b * T_ + t0 + lm) * H_;
        qa0.q[0] = ld16g(qr + hg * 16);
        qa0.q[1] = ld16g(qr + hg * 16 + 8);
        qa1.q[0] = ld16g(qr + 32 + hg * 16);
        qa1.q[1] = ld16g(qr + 32 + hg * 16 + 8);
    }

    v8f o0 = {}, o1 = {}, o2 = {}, o3 = {};
    float mrow[8], lrow[8];
#pragma unroll
    for (int r = 0; r < 8; ++r) { mrow[r] = -__builtin_inff(); lrow[r] = 0.0f; }

    // softmax in log2 domain: v_exp_f32 is natively exp2
    const float scale2 = 0.125f * 1.44269504088896340736f;  // H^-0.5 * log2(e)
    const int   nblk   = (t0 + 47) >> 5;

    const unsigned short* Kbase = kbuf  + (size_t)b * T_ * H_;
    const unsigned short* Vbase = vtbuf + (size_t)b * H_ * T_;
    unsigned short* pw = plds[wid];

    for (int blkI = 0; blkI < nblk; ++blkI) {
        const int s0 = blkI * 32;
        if (blkI + 1 < nblk) {                // speculative prefetch of next block
            __builtin_prefetch(Kbase + (size_t)(s0 + 32 + lm) * H_, 0, 0);
            __builtin_prefetch(Vbase + (size_t)lm * T_ + s0 + 32, 0, 0);
        }

        // ---- S(16x32) = Q(16x64) @ K_blk^T : two 16-col tiles, 2 WMMAs each ----
        v8f S0 = {}, S1 = {};
        {
            const unsigned short* kr = Kbase + (size_t)(s0 + lm) * H_;      // keys s0..s0+15
            FragBF kf;
            kf.q[0] = ld16g(kr + hg * 16);
            kf.q[1] = ld16g(kr + hg * 16 + 8);
            S0 = __builtin_amdgcn_wmma_f32_16x16x32_bf16(false, qa0.v, false, kf.v, (short)0, S0, false, false);
            kf.q[0] = ld16g(kr + 32 + hg * 16);
            kf.q[1] = ld16g(kr + 32 + hg * 16 + 8);
            S0 = __builtin_amdgcn_wmma_f32_16x16x32_bf16(false, qa1.v, false, kf.v, (short)0, S0, false, false);
        }
        {
            const unsigned short* kr = Kbase + (size_t)(s0 + 16 + lm) * H_; // keys s0+16..s0+31
            FragBF kf;
            kf.q[0] = ld16g(kr + hg * 16);
            kf.q[1] = ld16g(kr + hg * 16 + 8);
            S1 = __builtin_amdgcn_wmma_f32_16x16x32_bf16(false, qa0.v, false, kf.v, (short)0, S1, false, false);
            kf.q[0] = ld16g(kr + 32 + hg * 16);
            kf.q[1] = ld16g(kr + 32 + hg * 16 + 8);
            S1 = __builtin_amdgcn_wmma_f32_16x16x32_bf16(false, qa1.v, false, kf.v, (short)0, S1, false, false);
        }

        // ---- scale, causal mask, online softmax (rows live in 16-lane groups) ----
        float p0[8], p1[8], alpha[8];
#pragma unroll
        for (int r = 0; r < 8; ++r) {
            const int trow = t0 + r + hg * 8;
            float s0v = S0[r] * scale2;
            float s1v = S1[r] * scale2;
            if (s0 + lm > trow)      s0v = -__builtin_inff();
            if (s0 + 16 + lm > trow) s1v = -__builtin_inff();

            float rm = fmaxf(s0v, s1v);
#pragma unroll
            for (int d = 8; d >= 1; d >>= 1) rm = fmaxf(rm, __shfl_xor(rm, d, 32));
            const float mnew = fmaxf(mrow[r], rm);
            alpha[r] = exp2f(mrow[r] - mnew);
            mrow[r]  = mnew;
            const float e0 = exp2f(s0v - mnew);
            const float e1 = exp2f(s1v - mnew);
            float rs = e0 + e1;
#pragma unroll
            for (int d = 8; d >= 1; d >>= 1) rs += __shfl_xor(rs, d, 32);
            lrow[r] = lrow[r] * alpha[r] + rs;
            p0[r] = e0; p1[r] = e1;
        }

        // ---- D-layout P -> LDS -> A-layout fragment (per-wave scratch) ----
#pragma unroll
        for (int r = 0; r < 8; ++r) {
            const int m = r + hg * 8;
            pw[m * 32 + lm]      = f2bf(p0[r]);
            pw[m * 32 + 16 + lm] = f2bf(p1[r]);
        }
        asm volatile("s_wait_dscnt 0" ::: "memory");   // wave-local store->load ordering
        FragBF pa;
        pa.q[0] = *(const uint4*)(pw + lm * 32 + hg * 16);
        pa.q[1] = *(const uint4*)(pw + lm * 32 + hg * 16 + 8);

        // ---- O = diag(alpha) O + P(16x32) @ V_blk(32x64), 4 N-tiles ----
#pragma unroll
        for (int r = 0; r < 8; ++r) {
            o0[r] *= alpha[r]; o1[r] *= alpha[r];
            o2[r] *= alpha[r]; o3[r] *= alpha[r];
        }
        {
            FragBF vf;
            const unsigned short* vr;
            vr = Vbase + (size_t)(0  + lm) * T_ + s0 + hg * 16;
            vf.q[0] = ld16g(vr); vf.q[1] = ld16g(vr + 8);
            o0 = __builtin_amdgcn_wmma_f32_16x16x32_bf16(false, pa.v, false, vf.v, (short)0, o0, false, false);
            vr = Vbase + (size_t)(16 + lm) * T_ + s0 + hg * 16;
            vf.q[0] = ld16g(vr); vf.q[1] = ld16g(vr + 8);
            o1 = __builtin_amdgcn_wmma_f32_16x16x32_bf16(false, pa.v, false, vf.v, (short)0, o1, false, false);
            vr = Vbase + (size_t)(32 + lm) * T_ + s0 + hg * 16;
            vf.q[0] = ld16g(vr); vf.q[1] = ld16g(vr + 8);
            o2 = __builtin_amdgcn_wmma_f32_16x16x32_bf16(false, pa.v, false, vf.v, (short)0, o2, false, false);
            vr = Vbase + (size_t)(48 + lm) * T_ + s0 + hg * 16;
            vf.q[0] = ld16g(vr); vf.q[1] = ld16g(vr + 8);
            o3 = __builtin_amdgcn_wmma_f32_16x16x32_bf16(false, pa.v, false, vf.v, (short)0, o3, false, false);
        }
    }

    // ---- normalize by l and write f32 output ----
    float* O = out + ((size_t)b * T_ + t0) * H_;
#pragma unroll
    for (int r = 0; r < 8; ++r) {
        const float inv = 1.0f / lrow[r];
        float* orow = O + (size_t)(r + hg * 8) * H_;
        orow[lm]      = o0[r] * inv;
        orow[16 + lm] = o1[r] * inv;
        orow[32 + lm] = o2[r] * inv;
        orow[48 + lm] = o3[r] * inv;
    }
}

extern "C" void kernel_launch(void* const* d_in, const int* in_sizes, int n_in,
                              void* d_out, int out_size, void* d_ws, size_t ws_size,
                              hipStream_t stream) {
    (void)in_sizes; (void)n_in; (void)out_size; (void)ws_size;
    const float* x  = (const float*)d_in[0];
    const float* Wq = (const float*)d_in[1];
    const float* Wk = (const float*)d_in[2];
    const float* Wv = (const float*)d_in[3];

    // workspace layout (bf16 halves):
    //   qb  (B,T,H)  2 MB
    //   kb  (B,T,H)  2 MB
    //   vtb (B,H,T)  2 MB
    //   wtq/wtk/wtv (H,C) 128 KB each  -> total ~6.4 MB
    unsigned short* qb  = (unsigned short*)d_ws;
    unsigned short* kb  = qb  + (size_t)B_ * T_ * H_;
    unsigned short* vtb = kb  + (size_t)B_ * T_ * H_;
    unsigned short* wtq = vtb + (size_t)B_ * H_ * T_;
    unsigned short* wtk = wtq + (size_t)H_ * C_;
    unsigned short* wtv = wtk + (size_t)H_ * C_;
    float* out = (float*)d_out;

    wprep_kernel<<<(H_ * C_) / 256, 256, 0, stream>>>(Wq, Wk, Wv, wtq, wtk, wtv);
    qkv_proj_kernel<<<B_ * (T_ / 16), 128, 0, stream>>>(x, wtq, wtk, wtv, qb, kb, vtb);
    attn_kernel<<<(B_ * (T_ / 16)) / 4, 128, 0, stream>>>(qb, kb, vtb, out);
}
